// GRU_16234976378961
// MI455X (gfx1250) — compile-verified, matrix operand order checked by reading, not run on hardware
//
#include <hip/hip_runtime.h>
#include <hip/hip_bf16.h>

#define H   512
#define BSZ 16
#define T   2048
#define G3H (3 * H)      // 1536

typedef __bf16 bf16;
typedef __attribute__((ext_vector_type(16))) __bf16 v16bf;
typedef __attribute__((ext_vector_type(8)))  float  v8f;

union Frag { int4 i2[2]; v16bf v; };   // 32 bytes = 16 bf16 = 8 VGPRs

// ---------------------------------------------------------------------------
// Load a 16x32 bf16 WMMA operand fragment from a row-major matrix.
// ISA layout (CDNA5 ISA, 16-bit A-matrix 16x32): lane L holds row r = L&15;
//   elements e[0..7]  = M[r][k0 + (L>>4)*8 + j]
//   elements e[8..15] = M[r][k0 + 16 + (L>>4)*8 + j]
// B operand (32x16, B[k][n] = W[n][k]) uses the same pattern with n in the
// lane field, i.e. each lane reads 8 contiguous elements of a W row.
// ---------------------------------------------------------------------------
__device__ __forceinline__ v16bf load_frag(const bf16* __restrict__ base,
                                           int row0, int ldk, int k0, int lane) {
    int r    = lane & 15;
    int koff = (lane >> 4) * 8;
    const bf16* p = base + (size_t)(row0 + r) * ldk + k0 + koff;
    Frag f;
    f.i2[0] = *(const int4*)(p);
    f.i2[1] = *(const int4*)(p + 16);
    return f.v;
}

// ---------------------------------------------------------------------------
// fp32 -> bf16 conversion
// ---------------------------------------------------------------------------
__global__ void cvt_bf16(const float* __restrict__ src, bf16* __restrict__ dst, int n) {
    int i = blockIdx.x * blockDim.x + threadIdx.x;
    if (i < n) dst[i] = (bf16)src[i];
}

// ---------------------------------------------------------------------------
// Phase 1: x_gates[m, n] = sum_k x[m,k] * W_ih[n,k] + b_ih[n]
// M = B*T = 32768, N = 3H = 1536, K = H = 512.
// Block = 256 threads (8 waves) arranged 4(M) x 2(N) -> 128 x 128 block tile.
// Each wave computes a 32(M) x 64(N) tile: 2x4 v_wmma_f32_16x16x32_bf16
// accumulators, 8 WMMAs per 12 b128 loads per K-step (2 A + 4 B fragments).
// Operands (32 MB x + 1.5 MB W) are L2-resident; no LDS staging needed.
// ---------------------------------------------------------------------------
__global__ __launch_bounds__(256)
void gemm_xgates(const bf16* __restrict__ xbf, const bf16* __restrict__ wbf,
                 const float* __restrict__ bias, float* __restrict__ xg) {
    int lane = threadIdx.x & 31;
    int wave = threadIdx.x >> 5;                       // 0..7
    int m0 = blockIdx.x * 128 + (wave >> 1) * 32;      // 4 waves along M
    int n0 = blockIdx.y * 128 + (wave & 1)  * 64;      // 2 waves along N

    v8f c00 = {}, c01 = {}, c02 = {}, c03 = {};
    v8f c10 = {}, c11 = {}, c12 = {}, c13 = {};
    #pragma unroll 4
    for (int k = 0; k < H; k += 32) {
        v16bf a0 = load_frag(xbf, m0,      H, k, lane);
        v16bf a1 = load_frag(xbf, m0 + 16, H, k, lane);
        v16bf b0 = load_frag(wbf, n0,      H, k, lane);
        v16bf b1 = load_frag(wbf, n0 + 16, H, k, lane);
        v16bf b2 = load_frag(wbf, n0 + 32, H, k, lane);
        v16bf b3 = load_frag(wbf, n0 + 48, H, k, lane);
        c00 = __builtin_amdgcn_wmma_f32_16x16x32_bf16(false, a0, false, b0, (short)0, c00, false, false);
        c01 = __builtin_amdgcn_wmma_f32_16x16x32_bf16(false, a0, false, b1, (short)0, c01, false, false);
        c02 = __builtin_amdgcn_wmma_f32_16x16x32_bf16(false, a0, false, b2, (short)0, c02, false, false);
        c03 = __builtin_amdgcn_wmma_f32_16x16x32_bf16(false, a0, false, b3, (short)0, c03, false, false);
        c10 = __builtin_amdgcn_wmma_f32_16x16x32_bf16(false, a1, false, b0, (short)0, c10, false, false);
        c11 = __builtin_amdgcn_wmma_f32_16x16x32_bf16(false, a1, false, b1, (short)0, c11, false, false);
        c12 = __builtin_amdgcn_wmma_f32_16x16x32_bf16(false, a1, false, b2, (short)0, c12, false, false);
        c13 = __builtin_amdgcn_wmma_f32_16x16x32_bf16(false, a1, false, b3, (short)0, c13, false, false);
    }
    // C/D layout: lane -> N = lane&15, VGPR kk -> M = kk + 8*(lane>>4)
    int N  = lane & 15;
    int Mo = (lane >> 4) * 8;
    float bi0 = bias[n0 + N];
    float bi1 = bias[n0 + 16 + N];
    float bi2 = bias[n0 + 32 + N];
    float bi3 = bias[n0 + 48 + N];
    #pragma unroll
    for (int kk = 0; kk < 8; ++kk) {
        size_t r0 = (size_t)(m0 + Mo + kk)      * G3H + n0 + N;
        size_t r1 = (size_t)(m0 + 16 + Mo + kk) * G3H + n0 + N;
        xg[r0]      = c00[kk] + bi0;
        xg[r0 + 16] = c01[kk] + bi1;
        xg[r0 + 32] = c02[kk] + bi2;
        xg[r0 + 48] = c03[kk] + bi3;
        xg[r1]      = c10[kk] + bi0;
        xg[r1 + 16] = c11[kk] + bi1;
        xg[r1 + 32] = c12[kk] + bi2;
        xg[r1 + 48] = c13[kk] + bi3;
    }
}

// ---------------------------------------------------------------------------
// Phase 2: sequential GRU scan. One persistent 1024-thread workgroup
// (32 waves) on one WGP. h lives in LDS (f32 master + bf16 WMMA-A copy).
// Wave w owns N-tiles {w, 32+w, 64+w} => reset/update/new pre-activations for
// hidden columns [16w, 16w+16) stay inside the wave; only the h update needs
// workgroup barriers. W_hh (bf16, 1.5 MB) streams from L2 every step.
// ---------------------------------------------------------------------------
#define HBF_LD 528   // bf16 row pad: 1056 B, 16 B aligned, breaks bank stride

__global__ __launch_bounds__(1024)
void gru_scan(const float* __restrict__ xg, const bf16* __restrict__ whh,
              const float* __restrict__ bhh, float* __restrict__ out) {
    __shared__ __align__(16) float h32[BSZ][H];
    __shared__ __align__(16) bf16  hbf[BSZ][HBF_LD];

    int tid  = threadIdx.x;
    int lane = tid & 31;
    int wave = tid >> 5;                       // 0..31

    for (int i = tid; i < BSZ * H; i += 1024) {
        int rr = i >> 9, cc = i & (H - 1);
        h32[rr][cc] = 0.0f;
        hbf[rr][cc] = (bf16)0.0f;
    }
    __syncthreads();

    int N    = lane & 15;
    int koff = (lane >> 4) * 8;
    int Mo   = (lane >> 4) * 8;
    int col  = wave * 16 + N;                  // hidden column within each gate

    float bhr = bhh[col];
    float bhz = bhh[H + col];
    float bhn = bhh[2 * H + col];

    // W_hh rows for this lane's column, one per gate (B[k][n] = W_hh[n][k])
    const bf16* wp0 = whh + ((size_t)(0 * H + col)) * H + koff;
    const bf16* wp1 = whh + ((size_t)(1 * H + col)) * H + koff;
    const bf16* wp2 = whh + ((size_t)(2 * H + col)) * H + koff;

    for (int t = 0; t < T; ++t) {
        v8f cr = {}, cz = {}, cn = {};
        #pragma unroll 4
        for (int kc = 0; kc < H; kc += 32) {
            Frag fa;
            const bf16* ap = &hbf[lane & 15][kc + koff];
            fa.i2[0] = *(const int4*)(ap);
            fa.i2[1] = *(const int4*)(ap + 16);
            v16bf a = fa.v;

            Frag fb;
            fb.i2[0] = *(const int4*)(wp0 + kc);
            fb.i2[1] = *(const int4*)(wp0 + kc + 16);
            cr = __builtin_amdgcn_wmma_f32_16x16x32_bf16(false, a, false, fb.v, (short)0, cr, false, false);
            fb.i2[0] = *(const int4*)(wp1 + kc);
            fb.i2[1] = *(const int4*)(wp1 + kc + 16);
            cz = __builtin_amdgcn_wmma_f32_16x16x32_bf16(false, a, false, fb.v, (short)0, cz, false, false);
            fb.i2[0] = *(const int4*)(wp2 + kc);
            fb.i2[1] = *(const int4*)(wp2 + kc + 16);
            cn = __builtin_amdgcn_wmma_f32_16x16x32_bf16(false, a, false, fb.v, (short)0, cn, false, false);
        }
        __syncthreads();   // everyone done reading h (A fragments) this step

        #pragma unroll
        for (int kk = 0; kk < 8; ++kk) {
            int   mrow  = Mo + kk;                                   // batch row
            size_t xb   = ((size_t)mrow * T + t) * G3H + col;
            float gir   = xg[xb];
            float giz   = xg[xb + H];
            float gin   = xg[xb + 2 * H];
            float r_    = 1.0f / (1.0f + __expf(-(gir + cr[kk] + bhr)));
            float z_    = 1.0f / (1.0f + __expf(-(giz + cz[kk] + bhz)));
            float n_    = tanhf(gin + r_ * (cn[kk] + bhn));
            float hold  = h32[mrow][col];
            float hnew  = (1.0f - z_) * n_ + z_ * hold;
            out[((size_t)mrow * T + t) * H + col] = hnew;
            h32[mrow][col] = hnew;
            hbf[mrow][col] = (bf16)hnew;
        }
        __syncthreads();   // h fully updated before next step's GEMM
    }
}

// ---------------------------------------------------------------------------
// Launch. Workspace layout (offsets in bytes, all 256-aligned):
//   [0, 32 MB)        x in bf16
//   [+1.5 MB)         W_ih bf16
//   [+1.5 MB)         W_hh bf16
//   [+192 MB)         x_gates f32  (B*T x 3H)
// ---------------------------------------------------------------------------
extern "C" void kernel_launch(void* const* d_in, const int* in_sizes, int n_in,
                              void* d_out, int out_size, void* d_ws, size_t ws_size,
                              hipStream_t stream) {
    const float* x    = (const float*)d_in[0];
    const float* W_ih = (const float*)d_in[1];
    const float* W_hh = (const float*)d_in[2];
    const float* b_ih = (const float*)d_in[3];
    const float* b_hh = (const float*)d_in[4];
    float* out = (float*)d_out;

    char* ws = (char*)d_ws;
    bf16*  xbf   = (bf16*)(ws);                                   // 33,554,432 B
    bf16*  wihbf = (bf16*)(ws + 33554432);                        //  1,572,864 B
    bf16*  whhbf = (bf16*)(ws + 35127296);                        //  1,572,864 B
    float* xg    = (float*)(ws + 36700160);                       // 201,326,592 B

    int nx = BSZ * T * H;
    int nw = G3H * H;
    cvt_bf16<<<(nx + 255) / 256, 256, 0, stream>>>(x, xbf, nx);
    cvt_bf16<<<(nw + 255) / 256, 256, 0, stream>>>(W_ih, wihbf, nw);
    cvt_bf16<<<(nw + 255) / 256, 256, 0, stream>>>(W_hh, whhbf, nw);

    dim3 g1((BSZ * T) / 128, G3H / 128);   // (256, 12)
    gemm_xgates<<<g1, 256, 0, stream>>>(xbf, wihbf, b_ih, xg);

    gru_scan<<<1, 1024, 0, stream>>>(xg, whhbf, b_hh, out);
}
